// QuantizationLayer_10350871184036
// MI455X (gfx1250) — matrix operand order repaired and checked by reading, not compiled, not run on hardware
//
#include <hip/hip_runtime.h>
#include <hip/hip_bf16.h>

typedef __attribute__((ext_vector_type(16))) _Float16 v16h;
typedef __attribute__((ext_vector_type(8)))  float    v8f;

#define NEVENTS 250000
#define CBINS   9
#define HH      180
#define WW      240
#define BBATCH  8
#define IMG     640
#define VOXN    (2*CBINS*HH*WW*BBATCH)   // 6,220,800
#define HIDDEN  100
#define NTILES  7        // N padded to 112
#define KCHUNKS 4        // K padded to 128
#define FRAG_HALFS (NTILES*KCHUNKS*32*16) // 14336 halfs of pre-swizzled W2

__device__ __forceinline__ float leaky(float v) { return v > 0.f ? v : 0.1f * v; }

// ---------------- kernel 1: zero voxel grid + bmax ----------------
__global__ void zero_ws(float4* __restrict__ vox4, float* __restrict__ bmax) {
    long i = (long)blockIdx.x * blockDim.x + threadIdx.x;
    if (i < VOXN / 4) vox4[i] = make_float4(0.f, 0.f, 0.f, 0.f);
    if (i < BBATCH) bmax[i] = 0.f;
}

// ---------------- kernel 2: per-batch max of t (t > 0 -> uint-order trick) ----
__global__ void bmax_kernel(const float* __restrict__ ev, unsigned* __restrict__ bmax) {
    int e = blockIdx.x * blockDim.x + threadIdx.x;
    if (e >= NEVENTS) return;
    float t = ev[e * 5 + 2];
    int   b = (int)ev[e * 5 + 4];
    atomicMax(&bmax[b], __float_as_uint(t));
}

// ---------------- kernel 3: fused MLP (WMMA) + scatter-add --------------------
// grid = (ceil(N/128), CBINS), block = 256 (8 waves, 16 events per wave)
__global__ void __launch_bounds__(256)
mlp_scatter(const float* __restrict__ ev,
            const float* __restrict__ W1, const float* __restrict__ b1,
            const float* __restrict__ W2, const float* __restrict__ b2,
            const float* __restrict__ W3, const float* __restrict__ b3,
            const float* __restrict__ bmaxf, float* __restrict__ vox) {
    // W2 pre-swizzled into B-fragment order:
    // fragment f = nt*4+kc ; per lane 16 halfs contiguous -> 2x ds_load_b128
    __shared__ __align__(32) _Float16 sW2[FRAG_HALFS];
    __shared__ float sW1[HIDDEN], sB1[HIDDEN];
    __shared__ float sB2[NTILES * 16], sW3[NTILES * 16];
    __shared__ float sB3;
    __shared__ float rowbuf[8 * 16];

    const int tid = threadIdx.x;

    // ---- stage weights ----
    for (int i = tid; i < FRAG_HALFS; i += 256) {
        int f   = i >> 9;          // 512 halfs per fragment (32 lanes * 16)
        int rem = i & 511;
        int l   = rem >> 4;        // lane
        int h   = rem & 15;        // half within v16h
        int kc  = f & 3;
        int nt  = f >> 2;
        int vg  = h >> 1, pr = h & 1;
        // 16-bit B 32x16 layout (matches A): VGPR0..3 -> K=0..7, VGPR4..7 -> K=16..23 ; lanes 16-31: +8
        int k = kc * 32 + ((l & 16) ? 8 : 0) + ((vg < 4) ? 2 * vg : 8 + 2 * vg) + pr;
        int n = nt * 16 + (l & 15);
        float v = (k < HIDDEN && n < HIDDEN) ? W2[k * HIDDEN + n] : 0.f;
        sW2[i] = (_Float16)v;
    }
    for (int i = tid; i < HIDDEN; i += 256) { sW1[i] = W1[i]; sB1[i] = b1[i]; }
    for (int i = tid; i < NTILES * 16; i += 256) {
        sB2[i] = (i < HIDDEN) ? b2[i] : 0.f;
        sW3[i] = (i < HIDDEN) ? W3[i] : 0.f;
    }
    if (tid == 0) sB3 = b3[0];
    __syncthreads();

    const int wid  = tid >> 5;
    const int lane = tid & 31;
    const int lm   = lane & 15;
    const int bin  = blockIdx.y;

    const int ebase = (blockIdx.x * 8 + wid) * 16;
    const int eidx  = ebase + lm;
    const bool valid = (eidx < NEVENTS);
    const int e = valid ? eidx : (NEVENTS - 1);

    float tn = 0.f, xf = 0.f, yf = 0.f, pf = 0.f, bf = 0.f;
    if (lane < 16) {
        xf = ev[e * 5 + 0];
        yf = ev[e * 5 + 1];
        float t = ev[e * 5 + 2];
        pf = ev[e * 5 + 3];
        bf = ev[e * 5 + 4];
        tn = t / bmaxf[(int)bf];
    }
    float s = tn - (float)bin * (1.0f / (float)(CBINS - 1));

    // broadcast this lane's row scalar: A layout gives every lane M = lane&15
    float s_m = __shfl(s, lm, 32);

    // ---- layer 1 straight into A-fragment layout (16-bit A 16x32) ----
    // lanes 0-15: VGPR0..3 -> K=0..7, VGPR4..7 -> K=16..23 ; lanes 16-31: +8
    const int koff = (lane & 16) ? 8 : 0;
    v16h A[KCHUNKS];
#pragma unroll
    for (int kc = 0; kc < KCHUNKS; ++kc) {
#pragma unroll
        for (int h = 0; h < 16; ++h) {
            int vg = h >> 1, pr = h & 1;
            int kloc = ((vg < 4) ? 2 * vg : 8 + 2 * vg) + pr;
            int k = kc * 32 + koff + kloc;
            float v = 0.f;
            if (k < HIDDEN) v = leaky(fmaf(s_m, sW1[k], sB1[k]));
            A[kc][h] = (_Float16)v;
        }
    }

    // ---- layer 2 via WMMA, layer 3 folded into per-lane partial dot ----
    v8f facc = {0.f, 0.f, 0.f, 0.f, 0.f, 0.f, 0.f, 0.f};
    for (int nt = 0; nt < NTILES; ++nt) {
        int n = nt * 16 + lm;
        float binit = sB2[n];
        v8f acc = {binit, binit, binit, binit, binit, binit, binit, binit};
#pragma unroll
        for (int kc = 0; kc < KCHUNKS; ++kc) {
            const v16h Bf = *(const v16h*)&sW2[((nt * 4 + kc) * 32 + lane) * 16];
            acc = __builtin_amdgcn_wmma_f32_16x16x32_f16(
                false, A[kc], false, Bf, (short)0, acc, false, false);
        }
        float w3n = sW3[n];
#pragma unroll
        for (int r = 0; r < 8; ++r) facc[r] += leaky(acc[r]) * w3n;
    }

    // butterfly over the 16 N-lanes of each half (masks < 16 stay in-half)
#pragma unroll
    for (int m = 1; m < 16; m <<= 1) {
#pragma unroll
        for (int r = 0; r < 8; ++r) facc[r] += __shfl_xor(facc[r], m, 32);
    }

    // lane 0 holds rows 0..7, lane 16 holds rows 8..15 (C layout)
    if ((lane & 15) == 0) {
        int off = wid * 16 + ((lane >> 4) ? 8 : 0);
#pragma unroll
        for (int r = 0; r < 8; ++r) rowbuf[off + r] = facc[r];
    }
    __syncthreads();

    if (lane < 16 && valid) {
        float dot = rowbuf[wid * 16 + lane];
        float val = tn * (dot + sB3);
        // idx = x + W*y + W*H*bin + W*H*C*p + W*H*C*2*b  (all exact in f32, < 2^23)
        float fidx = xf + (float)WW * yf + (float)(WW * HH) * (float)bin +
                     (float)(WW * HH * CBINS) * pf +
                     (float)(WW * HH * CBINS * 2) * bf;
        int idx = (int)fidx;
        idx = idx < 0 ? 0 : (idx > VOXN - 1 ? VOXN - 1 : idx);
        atomicAdd(&vox[idx], val);
    }
}

// ---------------- kernel 4: bilinear letterbox to [8,18,640,640] --------------
// 180x240 -> 480x640 (ratio 0.375 both axes), top pad 80, fill 114.
// One thread = 4 output pixels (float4 store); HBM-write bound (~236 MB).
__global__ void resize_letterbox(const float* __restrict__ vox, float* __restrict__ out) {
    long gid = (long)blockIdx.x * blockDim.x + threadIdx.x;
    const long total = (long)BBATCH * 18 * IMG * (IMG / 4);
    if (gid >= total) return;
    int  xg = (int)(gid % (IMG / 4));
    long r  = gid / (IMG / 4);
    int  oy = (int)(r % IMG); r /= IMG;
    int  ch = (int)(r % 18);  r /= 18;
    int  b  = (int)r;
    long obase = ((long)(b * 18 + ch) * IMG + oy) * IMG + (long)xg * 4;

    float4 o;
    if (oy < 80 || oy >= 560) {
        o = make_float4(114.f, 114.f, 114.f, 114.f);
    } else {
        int p = ch / CBINS, c = ch % CBINS;
        const float* src = vox + (long)((b * 2 + p) * CBINS + c) * (HH * WW);
        float sy = ((float)(oy - 80) + 0.5f) * 0.375f - 0.5f;
        int   iy = (int)floorf(sy);
        float fy = sy - (float)iy;
        int y0 = min(max(iy, 0), HH - 1);
        int y1 = min(max(iy + 1, 0), HH - 1);
        float res[4];
#pragma unroll
        for (int j = 0; j < 4; ++j) {
            int   ox = xg * 4 + j;
            float sx = ((float)ox + 0.5f) * 0.375f - 0.5f;
            int   ix = (int)floorf(sx);
            float fx = sx - (float)ix;
            int x0 = min(max(ix, 0), WW - 1);
            int x1 = min(max(ix + 1, 0), WW - 1);
            float v00 = src[y0 * WW + x0], v01 = src[y0 * WW + x1];
            float v10 = src[y1 * WW + x0], v11 = src[y1 * WW + x1];
            float top = v00 + (v01 - v00) * fx;
            float bot = v10 + (v11 - v10) * fx;
            res[j] = top + (bot - top) * fy;
        }
        o = make_float4(res[0], res[1], res[2], res[3]);
    }
    *(float4*)(out + obase) = o;
}

extern "C" void kernel_launch(void* const* d_in, const int* in_sizes, int n_in,
                              void* d_out, int out_size, void* d_ws, size_t ws_size,
                              hipStream_t stream) {
    const float* ev = (const float*)d_in[0];
    const float* W1 = (const float*)d_in[1];
    const float* b1 = (const float*)d_in[2];
    const float* W2 = (const float*)d_in[3];
    const float* b2 = (const float*)d_in[4];
    const float* W3 = (const float*)d_in[5];
    const float* b3 = (const float*)d_in[6];

    float* bmax = (float*)d_ws;                       // 8 floats
    float* vox  = (float*)((char*)d_ws + 256);        // 6,220,800 floats
    float* out  = (float*)d_out;                      // [8,18,640,640]

    zero_ws<<<(VOXN / 4 + 255) / 256, 256, 0, stream>>>((float4*)vox, bmax);
    bmax_kernel<<<(NEVENTS + 255) / 256, 256, 0, stream>>>(ev, (unsigned*)bmax);

    dim3 grid((NEVENTS + 127) / 128, CBINS);
    mlp_scatter<<<grid, 256, 0, stream>>>(ev, W1, b1, W2, b2, W3, b3, bmax, vox);

    long total4 = (long)BBATCH * 18 * IMG * (IMG / 4);
    resize_letterbox<<<(int)((total4 + 255) / 256), 256, 0, stream>>>(vox, out);
}
